// HILNet_47416438948429
// MI455X (gfx1250) — matrix operand order repaired and checked by reading, not compiled, not run on hardware
//
#include <hip/hip_runtime.h>
#include <math.h>

#define C_H 128
#define D_CNT 9
#define MTILES 8            // 8 x 16 = 128 rows per GEMM block (amortize W staging)

typedef __attribute__((ext_vector_type(16))) __bf16 v16bf;
typedef __attribute__((ext_vector_type(8)))  float  v8f;
typedef __attribute__((ext_vector_type(4)))  unsigned v4u;
typedef __attribute__((ext_vector_type(8)))  unsigned v8u;

union FragU { uint4 q[2]; v16bf v; };

// fast silu: v * rcp(1+exp(-v)) -> v_exp_f32 + v_rcp_f32
__device__ __forceinline__ float fast_silu(float v) {
    return v * __builtin_amdgcn_rcpf(1.0f + __expf(-v));
}

// LDS byte offset of a __shared__ object = low 32 bits of its generic VA
// (LDS aperture: addr[31:0] is the offset within the workgroup's LDS).
__device__ __forceinline__ unsigned lds_byte_off(const void* p) {
    return (unsigned)(uintptr_t)p;
}

// Issue a 1-D TDM copy: nW8 x 8-byte words, global -> LDS (D# per ISA 8.3/8.4).
// Per-wave instruction; tracked by TENSORcnt.
__device__ __forceinline__ void tdm_load_1d(unsigned ldsOff, const void* gptr, unsigned nW8) {
    const unsigned long long ga = (unsigned long long)(uintptr_t)gptr;
    v4u g0;
    g0.x = 1u;                                    // count=1 (valid), user mode
    g0.y = ldsOff;                                // lds_addr (bytes)
    g0.z = (unsigned)ga;                          // global_addr[31:0]
    g0.w = (unsigned)((ga >> 32) & 0x01FFFFFFu)   // global_addr[56:32]
         | (2u << 30);                            // type=2 ("image")
    v8u g1 = { (3u << 16),                        // data_size=3 (8 bytes)
               (nW8 & 0xFFFFu) << 16,             // tensor_dim0[15:0]
               (nW8 >> 16) | (1u << 16),          // tensor_dim0[31:16] | tensor_dim1=1
               (nW8 & 0xFFFFu) << 16,             // tensor_dim1 hi=0 | tile_dim0
               0u,                                // tile_dim1=0, tile_dim2=0 (1-D)
               nW8,                               // tensor_dim0_stride lo
               0u, 0u };
    v4u g2 = { 0u, 0u, 0u, 0u };
    v4u g3 = { 0u, 0u, 0u, 0u };
    asm volatile("tensor_load_to_lds %0, %1, %2, %3"
                 :: "s"(g0), "s"(g1), "s"(g2), "s"(g3)
                 : "memory");
}

// A-matrix 16x32 bf16 fragment (CDNA5 ISA 7.12.2):
// lanes 0-15 (M=lane): K = b..b+7 and b+16..b+23 (b=0); lanes 16-31: b=8.
__device__ __forceinline__ v16bf load_a_frag(const __bf16* row, int kb, int laneHi) {
    FragU u;
    const uint4* p = reinterpret_cast<const uint4*>(row + kb * 32 + laneHi * 8);
    u.q[0] = p[0];
    u.q[1] = p[2];   // +16 bf16 elements
    return u.v;
}

// B-matrix 32x16 bf16 fragment: lanes 0-15 (N=lane) hold K=0..7 and 16..23,
// lanes 16-31 hold K=8..15 and 24..31. Read from LDS-resident W^T rows (2x b128).
__device__ __forceinline__ v16bf load_b_frag(const __bf16* row, int kb, int laneHi) {
    FragU u;
    const uint4* p = reinterpret_cast<const uint4*>(row + kb * 32 + laneHi * 8);
    u.q[0] = p[0];
    u.q[1] = p[2];
    return u.v;
}

// One-time per layer: W[K][128] fp32 -> W^T[128][K+8] bf16 (pad zeroed).
template <int K>
__global__ __launch_bounds__(256) void prep_wt_kernel(
    const float* __restrict__ W, __bf16* __restrict__ Wt)
{
    constexpr int AP = K + 8;
    const int idx = blockIdx.x * 256 + threadIdx.x;
    if (idx >= C_H * AP) return;
    const int n = idx / AP, k = idx % AP;
    Wt[idx] = (k < K) ? (__bf16)W[k * C_H + n] : (__bf16)0.0f;
}

// Block = 128 rows (8 tiles of 16): W^T staged once via TDM, then per tile:
// stage A (VALU fp32->bf16, overlapped with TDM on tile 0), 16x16x32 WMMAs,
// fused epilogue.
// MODE 0: silu(acc + bias)                       -> out[r][ch]
// MODE 1: BN-affine(LeakyReLU(acc + bias))       -> out[r][ch] (in-place safe)
// MODE 2: like MODE 1 but atomically pooled into -> out[batch[r]][ch]
template <int K, int MODE>
__global__ __launch_bounds__(256) void gemm_epilogue_kernel(
    const float* __restrict__ A0, const float* __restrict__ A1,
    const __bf16* __restrict__ Wt, const float* __restrict__ bias,
    const float* __restrict__ g,   const float* __restrict__ be,
    const int*   __restrict__ batch,
    float* __restrict__ out, int nrows)
{
    constexpr int AP = K + 8;                 // padded rows: conflict-free LDS
    __shared__ __align__(16) __bf16 ldsA[16  * AP];
    __shared__ __align__(16) __bf16 ldsB[C_H * AP];

    const int tid    = threadIdx.x;
    const int wave   = tid >> 5;
    const int lane   = tid & 31;
    const int laneHi = lane >> 4;
    const int nIdx   = lane & 15;
    const int chBase = wave * 16;             // 8 waves x 16 = 128 channels

    // Each wave DMAs its own 16-row slice of W^T into LDS (16*AP bf16).
    {
        const unsigned sliceBytes = 16u * AP * 2u;
        const unsigned nW8        = sliceBytes / 8u;
        tdm_load_1d(lds_byte_off(ldsB) + wave * sliceBytes,
                    (const char*)Wt + wave * sliceBytes, nW8);
    }

    const int blockBase = blockIdx.x * (16 * MTILES);

    for (int t = 0; t < MTILES; ++t) {
        const int base = blockBase + t * 16;
        if (base >= nrows) break;             // uniform

        // Stage A tile (h (+agg)) as bf16; on t==0 this overlaps the TDM.
        for (int idx = tid; idx < 16 * K; idx += 256) {
            const int m = idx / K, k = idx % K;
            const int r = base + m;
            float v = 0.0f;
            if (r < nrows) {
                v = A0[(size_t)r * K + k];
                if (MODE != 0) v += A1[(size_t)r * K + k];
            }
            ldsA[m * AP + k] = (__bf16)v;
        }
        if (t == 0) __builtin_amdgcn_s_wait_tensorcnt(0);
        __syncthreads();

        const __bf16* aRow = ldsA + nIdx * AP;
        const __bf16* bRow = ldsB + (chBase + nIdx) * AP;

        v8f acc = {};
#pragma unroll
        for (int kb = 0; kb < K / 32; ++kb) {
            const v16bf a = load_a_frag(aRow, kb, laneHi);
            const v16bf b = load_b_frag(bRow, kb, laneHi);
            acc = __builtin_amdgcn_wmma_f32_16x16x32_bf16(
                false, a, false, b, (short)0, acc, false, false);
        }

        // Epilogue. D layout: lane n = lane&15, rows m = j + 8*(lane>>4).
        const int   ch = chBase + nIdx;
        const float bj = bias[ch];
        float gm = 0.0f, bem = 0.0f;
        if (MODE != 0) { gm = g[ch] * rsqrtf(1.0f + 1e-5f); bem = be[ch]; }

        float y[8];
#pragma unroll
        for (int j = 0; j < 8; ++j) {
            float v = acc[j] + bj;
            if (MODE == 0) {
                y[j] = fast_silu(v);
            } else {
                v = v > 0.0f ? v : 0.01f * v;          // leaky relu
                y[j] = gm * v + bem;                   // frozen-BN affine
            }
        }

        const int rowBase = base + 8 * laneHi;
        if (base + 16 <= nrows) {                      // full tile: branch-free
#pragma unroll
            for (int j = 0; j < 8; ++j) {
                const int r = rowBase + j;
                if (MODE == 2) unsafeAtomicAdd(out + (size_t)batch[r] * C_H + ch, y[j]);
                else           out[(size_t)r * C_H + ch] = y[j];
            }
        } else {                                       // ragged tail
#pragma unroll
            for (int j = 0; j < 8; ++j) {
                const int r = rowBase + j;
                if (r < nrows) {
                    if (MODE == 2) unsafeAtomicAdd(out + (size_t)batch[r] * C_H + ch, y[j]);
                    else           out[(size_t)r * C_H + ch] = y[j];
                }
            }
        }
        __syncthreads();                               // before next A overwrite
    }
}

// Fused edge phase: one wave32 per edge. RBF -> radial(silu) -> h[row]*radial
// -> hardware f32 atomic scatter into agg[col]. Never materializes [E,128].
__global__ __launch_bounds__(256) void edge_kernel(
    const float* __restrict__ h,   const float* __restrict__ pos,
    const int*   __restrict__ row, const int*   __restrict__ col,
    const float* __restrict__ Wc,  const float* __restrict__ bc,
    float* __restrict__ agg, int nEdges)
{
    __shared__ float sWc[D_CNT * C_H];
    __shared__ float sBc[C_H];
    const int tid = threadIdx.x;
    for (int i = tid; i < D_CNT * C_H; i += 256) sWc[i] = Wc[i];
    if (tid < C_H) sBc[tid] = bc[tid];
    __syncthreads();

    const int lane   = tid & 31;
    const int waveId = (blockIdx.x * 256 + tid) >> 5;
    const int nWaves = (gridDim.x * 256) >> 5;
    const float sigInv = 1.5f;                 // 1 / (6/9)

    for (int e = waveId; e < nEdges; e += nWaves) {
        const int r = row[e];
        const int c = col[e];
        const float dx = pos[r * 3 + 0] - pos[c * 3 + 0];
        const float dy = pos[r * 3 + 1] - pos[c * 3 + 1];
        const float dz = pos[r * 3 + 2] - pos[c * 3 + 2];
        const float d  = sqrtf(dx * dx + dy * dy + dz * dz);

        float rbf[D_CNT];
#pragma unroll
        for (int k = 0; k < D_CNT; ++k) {
            const float t = (d - 0.75f * (float)k) * sigInv;   // mu_k = 6k/8
            rbf[k] = __expf(-t * t);
        }

        const int c0 = lane * 4;               // wave covers all 128 channels
        const float4 hv = *reinterpret_cast<const float4*>(h + (size_t)r * C_H + c0);
        const float hArr[4] = {hv.x, hv.y, hv.z, hv.w};
#pragma unroll
        for (int u = 0; u < 4; ++u) {
            const int ch = c0 + u;
            float a = sBc[ch];
#pragma unroll
            for (int k = 0; k < D_CNT; ++k) a = fmaf(rbf[k], sWc[k * C_H + ch], a);
            unsafeAtomicAdd(agg + (size_t)c * C_H + ch, hArr[u] * fast_silu(a));
        }
    }
}

__global__ __launch_bounds__(256) void zero4_kernel(float4* __restrict__ p, long n4) {
    const long i = (long)blockIdx.x * 256 + threadIdx.x;
    if (i < n4) p[i] = make_float4(0.f, 0.f, 0.f, 0.f);
}

extern "C" void kernel_launch(void* const* d_in, const int* in_sizes, int n_in,
                              void* d_out, int out_size, void* d_ws, size_t ws_size,
                              hipStream_t stream)
{
    const float* x     = (const float*)d_in[0];
    const float* pos   = (const float*)d_in[1];
    const int*   eidx  = (const int*)  d_in[2];
    const int*   batch = (const int*)  d_in[3];
    const float* W_lin = (const float*)d_in[4];
    const float* b_lin = (const float*)d_in[5];

    const int nNodes = in_sizes[0] / 64;
    const int nEdges = in_sizes[2] / 2;
    const int* row = eidx;            // edge_index[0]
    const int* col = eidx + nEdges;   // edge_index[1]

    float*  hBuf = (float*)d_ws;                      // [nNodes,128] fp32
    float*  agg  = hBuf + (size_t)nNodes * C_H;       // [nNodes,128] fp32
    __bf16* Wt   = (__bf16*)(agg + (size_t)nNodes * C_H);  // [128][K+8] bf16
    const long nodeElems = (long)nNodes * C_H;

    const int gemmBlocks = (nNodes + 16 * MTILES - 1) / (16 * MTILES);
    const int zeroBlocks = (int)((nodeElems / 4 + 255) / 256);

    // h = silu(x @ W_lin + b_lin)   (K=64, 2 WMMAs per tile)
    prep_wt_kernel<64><<<(C_H * 72 + 255) / 256, 256, 0, stream>>>(W_lin, Wt);
    gemm_epilogue_kernel<64, 0><<<gemmBlocks, 256, 0, stream>>>(
        x, nullptr, Wt, b_lin, nullptr, nullptr, nullptr, hBuf, nNodes);

    for (int l = 0; l < 3; ++l) {
        const float* Wc = (const float*)d_in[6  + 6 * l];
        const float* bc = (const float*)d_in[7  + 6 * l];
        const float* Wn = (const float*)d_in[8  + 6 * l];
        const float* bn = (const float*)d_in[9  + 6 * l];
        const float* g  = (const float*)d_in[10 + 6 * l];
        const float* be = (const float*)d_in[11 + 6 * l];

        zero4_kernel<<<zeroBlocks, 256, 0, stream>>>((float4*)agg, nodeElems / 4);
        edge_kernel<<<2048, 256, 0, stream>>>(hBuf, pos, row, col, Wc, bc, agg, nEdges);

        prep_wt_kernel<128><<<(C_H * 136 + 255) / 256, 256, 0, stream>>>(Wn, Wt);
        if (l < 2) {
            // h = BN(LeakyReLU((h + agg) @ Wn + bn))  -- in-place
            gemm_epilogue_kernel<128, 1><<<gemmBlocks, 256, 0, stream>>>(
                hBuf, agg, Wt, bn, g, be, nullptr, hBuf, nNodes);
        } else {
            // final layer: fuse global_add_pool -> out[batch[r]] += y
            zero4_kernel<<<(out_size / 4 + 255) / 256, 256, 0, stream>>>(
                (float4*)d_out, (long)out_size / 4);
            gemm_epilogue_kernel<128, 2><<<gemmBlocks, 256, 0, stream>>>(
                hBuf, agg, Wt, bn, g, be, batch, (float*)d_out, nNodes);
        }
    }
}
